// BinConv2d_45887430590599
// MI455X (gfx1250) — 8-bit matrix operand order repaired and checked by reading, compiled, not run on hardware
//
#include <hip/hip_runtime.h>
#include <cstdint>

#define C_IN  256
#define HH    56
#define WW    56
#define NN    32
#define OO    256
#define KTOT  2304
#define HWSZ  3136
#define HP    58
#define WP    58

typedef int v8i __attribute__((ext_vector_type(8)));

union B8frag { uint4 q[2]; v8i v; };

__global__ __launch_bounds__(256) void zero_ws_kernel(uint4* __restrict__ p)
{
    const size_t i = (size_t)blockIdx.x * 256 + threadIdx.x;
    p[i] = make_uint4(0u, 0u, 0u, 0u);
}

__global__ __launch_bounds__(256) void bn_stats_kernel(
    const float* __restrict__ x, const float* __restrict__ gamma,
    const float* __restrict__ beta, float* __restrict__ bn_scale,
    float* __restrict__ bn_shift)
{
    const int c = blockIdx.x;
    const int t = threadIdx.x;
    float s = 0.f, s2 = 0.f;
    for (int n = 0; n < NN; ++n) {
        const float* p = x + ((size_t)n * C_IN + c) * HWSZ;
        for (int i = t; i < HWSZ; i += 256) {
            float v = p[i];
            s += v; s2 += v * v;
        }
    }
    __shared__ float rs[256], rq[256];
    rs[t] = s; rq[t] = s2;
    __syncthreads();
    for (int st = 128; st > 0; st >>= 1) {
        if (t < st) { rs[t] += rs[t + st]; rq[t] += rq[t + st]; }
        __syncthreads();
    }
    if (t == 0) {
        const float cnt  = (float)(NN * HWSZ);
        const float mean = rs[0] / cnt;
        const float var  = rq[0] / cnt - mean * mean;
        const float r    = rsqrtf(var + 1e-4f);
        const float sc   = gamma[c] * r;
        bn_scale[c] = sc;
        bn_shift[c] = beta[c] - mean * sc;
    }
}

__global__ __launch_bounds__(256) void bin_act_kernel(
    const float* __restrict__ x, const float* __restrict__ bn_scale,
    const float* __restrict__ bn_shift, int8_t* __restrict__ xb)
{
    const size_t idx = (size_t)blockIdx.x * 256 + threadIdx.x;
    const int hw = (int)(idx % HWSZ);
    const int c  = (int)((idx / HWSZ) % C_IN);
    const int n  = (int)(idx / ((size_t)HWSZ * C_IN));
    const int hh = hw / WW, w = hw % WW;
    const float v = x[idx] * bn_scale[c] + bn_shift[c];
    const int8_t s = (v > 0.f) ? (int8_t)1 : ((v < 0.f) ? (int8_t)-1 : (int8_t)0);
    xb[(((size_t)n * HP + (hh + 1)) * WP + (w + 1)) * C_IN + c] = s;
}

__global__ __launch_bounds__(256) void w_prep_kernel(
    const float* __restrict__ w, int8_t* __restrict__ wb,
    float* __restrict__ wscale)
{
    const int o = blockIdx.x;
    const int c = threadIdx.x;
    __shared__ float tapsum[9];
    __shared__ float red[256];
    if (c < 9) tapsum[c] = 0.f;
    __syncthreads();

    float wv[9];
    const float* wp = w + ((size_t)o * C_IN + c) * 9;
    #pragma unroll
    for (int k = 0; k < 9; ++k) {
        wv[k] = wp[k];
        atomicAdd(&tapsum[k], wv[k]);
    }
    __syncthreads();

    float absacc = 0.f;
    #pragma unroll
    for (int k = 0; k < 9; ++k) {
        const float mean = tapsum[k] * (1.f / 256.f);
        float cv = wv[k] - mean;
        cv = fminf(fmaxf(cv, -1.f), 1.f);
        absacc += fabsf(cv);
        const int8_t s = (cv > 0.f) ? (int8_t)1 : ((cv < 0.f) ? (int8_t)-1 : (int8_t)0);
        wb[(size_t)o * KTOT + k * C_IN + c] = s;
    }
    red[c] = absacc;
    __syncthreads();
    for (int st = 128; st > 0; st >>= 1) {
        if (c < st) red[c] += red[c + st];
        __syncthreads();
    }
    if (c == 0) wscale[o] = red[0] * (1.f / (float)KTOT);
}

__global__ __launch_bounds__(256) void bconv_wmma_kernel(
    const int8_t* __restrict__ xb,
    const int8_t* __restrict__ wb,
    const float* __restrict__ wscale,
    const float* __restrict__ bias,
    float* __restrict__ out)
{
    __shared__ uint4 lds_abuf[2][512];
    __shared__ float lds_o[OO * 32];

    const int bx = blockIdx.x;
    const int wt = bx & 3;
    const int ht = (bx >> 2) % (HH / 2);
    const int n  = bx / (4 * (HH / 2));
    const int h0 = ht * 2;
    const int w0 = wt * 16;

    const int t    = threadIdx.x;
    const int lane = t & 31;
    const int wave = t >> 5;
    const int hi   = lane >> 4;
    const int m    = lane & 15;

    const int oc0 = wave * 32 + m;
    const int oc1 = oc0 + 16;

    v8i acc00 = {0,0,0,0,0,0,0,0};
    v8i acc01 = {0,0,0,0,0,0,0,0};
    v8i acc10 = {0,0,0,0,0,0,0,0};
    v8i acc11 = {0,0,0,0,0,0,0,0};

    const int a_base0 = m * 64 + hi * 4;
    const int a_base1 = (16 + m) * 64 + hi * 4;

    const int sp = t >> 3;
    const int sg = t & 7;
    const int spr = sp >> 4;
    const int spw = sp & 15;

    auto stage = [&](int tap, int buf) {
        const int kh = tap / 3, kw = tap % 3;
        const size_t srow = ((size_t)n * HP + (h0 + spr + kh)) * WP + (w0 + spw + kw);
        const unsigned long long ga =
            (unsigned long long)(uintptr_t)(xb + srow * C_IN + sg * 32);
        const unsigned ldst =
            (unsigned)(uintptr_t)(&lds_abuf[buf][0]) + (unsigned)(sp * 256 + sg * 32);
        asm volatile("global_load_async_to_lds_b128 %0, %1, off"
                     :: "v"(ldst), "v"(ga) : "memory");
        asm volatile("global_load_async_to_lds_b128 %0, %1, off offset:16"
                     :: "v"(ldst), "v"(ga) : "memory");
    };

    stage(0, 0);
    asm volatile("s_wait_asynccnt 0x0" ::: "memory");
    __syncthreads();

    for (int tap = 0; tap < 9; ++tap) {
        if (tap < 8) stage(tap + 1, (tap + 1) & 1);

        const unsigned* lds_a = (const unsigned*)&lds_abuf[tap & 1][0];
        const size_t tap_byte = (size_t)tap * C_IN;
        #pragma unroll
        for (int kc = 0; kc < 4; ++kc) {
            const unsigned* ap0 = lds_a + a_base0 + kc * 16;
            const unsigned* ap1 = lds_a + a_base1 + kc * 16;
            v8i a0, a1;
            a0[0]=(int)ap0[0];  a0[1]=(int)ap0[1];  a0[2]=(int)ap0[2];  a0[3]=(int)ap0[3];
            a0[4]=(int)ap0[8];  a0[5]=(int)ap0[9];  a0[6]=(int)ap0[10]; a0[7]=(int)ap0[11];
            a1[0]=(int)ap1[0];  a1[1]=(int)ap1[1];  a1[2]=(int)ap1[2];  a1[3]=(int)ap1[3];
            a1[4]=(int)ap1[8];  a1[5]=(int)ap1[9];  a1[6]=(int)ap1[10]; a1[7]=(int)ap1[11];

            const size_t kbyte = tap_byte + (size_t)kc * 64 + (size_t)hi * 16;
            const uint4* b0p = (const uint4*)(wb + (size_t)oc0 * KTOT + kbyte);
            const uint4* b1p = (const uint4*)(wb + (size_t)oc1 * KTOT + kbyte);
            B8frag b0, b1;
            b0.q[0] = b0p[0]; b0.q[1] = b0p[2];
            b1.q[0] = b1p[0]; b1.q[1] = b1p[2];

            acc00 = __builtin_amdgcn_wmma_i32_16x16x64_iu8(true, a0, true, b0.v, acc00, false, false);
            acc01 = __builtin_amdgcn_wmma_i32_16x16x64_iu8(true, a0, true, b1.v, acc01, false, false);
            acc10 = __builtin_amdgcn_wmma_i32_16x16x64_iu8(true, a1, true, b0.v, acc10, false, false);
            acc11 = __builtin_amdgcn_wmma_i32_16x16x64_iu8(true, a1, true, b1.v, acc11, false, false);
        }

        if (tap < 8) {
            asm volatile("s_wait_asynccnt 0x0" ::: "memory");
            __syncthreads();
        }
    }

    {
        const float sc0 = wscale[oc0], bi0 = bias[oc0];
        const float sc1 = wscale[oc1], bi1 = bias[oc1];
        #pragma unroll
        for (int r = 0; r < 8; ++r) {
            const int mm0 = r + hi * 8;
            const int mm1 = 16 + mm0;
            lds_o[oc0 * 32 + mm0] = fmaxf(((float)acc00[r] + bi0) * sc0, 0.f);
            lds_o[oc1 * 32 + mm0] = fmaxf(((float)acc01[r] + bi1) * sc1, 0.f);
            lds_o[oc0 * 32 + mm1] = fmaxf(((float)acc10[r] + bi0) * sc0, 0.f);
            lds_o[oc1 * 32 + mm1] = fmaxf(((float)acc11[r] + bi1) * sc1, 0.f);
        }
    }
    __syncthreads();

    #pragma unroll
    for (int i = 0; i < 32; ++i) {
        const int flat = i * 256 + t;
        const int o  = flat >> 5;
        const int mm = flat & 31;
        const int pr = mm >> 4;
        const int w  = w0 + (mm & 15);
        if (w < WW)
            out[(((size_t)n * OO + o) * HH + (h0 + pr)) * WW + w] = lds_o[flat];
    }
}

extern "C" void kernel_launch(void* const* d_in, const int* in_sizes, int n_in,
                              void* d_out, int out_size, void* d_ws, size_t ws_size,
                              hipStream_t stream) {
    (void)in_sizes; (void)n_in; (void)out_size; (void)ws_size;
    const float* x      = (const float*)d_in[0];
    const float* gamma  = (const float*)d_in[1];
    const float* beta   = (const float*)d_in[2];
    const float* weight = (const float*)d_in[3];
    const float* bias   = (const float*)d_in[4];
    float* out = (float*)d_out;

    char* ws = (char*)d_ws;
    float*  bn_scale = (float*)(ws + 0);
    float*  bn_shift = (float*)(ws + 1024);
    float*  wscale   = (float*)(ws + 2048);
    int8_t* wb       = (int8_t*)(ws + 4096);
    int8_t* xb       = (int8_t*)(ws + (1u << 20));

    const size_t pad_bytes = (size_t)NN * HP * WP * C_IN;
    zero_ws_kernel<<<(unsigned)(pad_bytes / 16 / 256), 256, 0, stream>>>((uint4*)xb);

    bn_stats_kernel<<<C_IN, 256, 0, stream>>>(x, gamma, beta, bn_scale, bn_shift);

    const size_t total = (size_t)NN * C_IN * HWSZ;
    bin_act_kernel<<<(unsigned)(total / 256), 256, 0, stream>>>(x, bn_scale, bn_shift, xb);

    w_prep_kernel<<<OO, 256, 0, stream>>>(weight, wb, wscale);

    bconv_wmma_kernel<<<NN * (HH / 2) * 4, 256, 0, stream>>>(xb, wb, wscale, bias, out);
}
